// DuplicationRemovalNetwork_87832081203241
// MI455X (gfx1250) — matrix operand order repaired and checked by reading, not compiled, not run on hardware
//
#include <hip/hip_runtime.h>

// Problem constants (from reference setup_inputs)
#define KP 2048   // proposals
#define NS 18     // size classes
#define NC 19     // semantic classes
#define DD 1024   // feature dim
#define DF 128    // embedding dim
#define NH 16     // heads
#define DK 64     // head dim

typedef __attribute__((ext_vector_type(16))) _Float16 v16h;
typedef __attribute__((ext_vector_type(8)))  float    v8f;
typedef __attribute__((ext_vector_type(4)))  int      v4i;

// ---------------------------------------------------------------------------
// CDNA5 async global->LDS staging (ASYNCcnt path), with safe fallbacks
// ---------------------------------------------------------------------------
#if __has_builtin(__builtin_amdgcn_global_load_async_to_lds_b128)
#define HAS_ASYNC_LDS 1
#else
#define HAS_ASYNC_LDS 0
#endif

// exact parameter types: v4i in global (AS1) / LDS (AS3) address spaces
typedef __attribute__((address_space(1))) v4i gv4i;
typedef __attribute__((address_space(3))) v4i lv4i;

// low 32 bits of a generic LDS pointer == LDS byte offset (aperture rule)
__device__ __forceinline__ unsigned lds_off(const void* p) {
  return (unsigned)(unsigned long long)p;
}

// copy 16 bytes global -> LDS
__device__ __forceinline__ void cp16_g2l(const _Float16* g, _Float16* l) {
#if HAS_ASYNC_LDS
  __builtin_amdgcn_global_load_async_to_lds_b128(
      (gv4i*)(unsigned long long)g,
      (lv4i*)(unsigned long long)lds_off(l), /*offset=*/0, /*cpol=*/0);
#else
  *(uint4*)l = *(const uint4*)g;
#endif
}

#if HAS_ASYNC_LDS
#if __has_builtin(__builtin_amdgcn_s_wait_asynccnt)
#define WAIT_ASYNC(n) do { __builtin_amdgcn_s_wait_asynccnt(n); \
                           asm volatile("" ::: "memory"); } while (0)
#else
#define WAIT_ASYNC(n) asm volatile("s_wait_asynccnt %0" :: "i"(n) : "memory")
#endif
#else
#define WAIT_ASYNC(n) asm volatile("" ::: "memory")
#endif

// ---------------------------------------------------------------------------
// WMMA helpers (V_WMMA_F32_16X16X32_F16, wave32)
// ---------------------------------------------------------------------------
__device__ __forceinline__ v8f wmma16(v16h a, v16h b, v8f c) {
  return __builtin_amdgcn_wmma_f32_16x16x32_f16(
      /*neg_a=*/false, a, /*neg_b=*/false, b,
      /*c_mod=*/(short)0, c, /*reuse_a=*/false, /*reuse_b=*/false);
}

// A-matrix fragment: 16x32 f16, row-major source, leading dim lda.
// lane L holds row (L%16); element e -> K = kk0 + (L/16)*16 + e
__device__ __forceinline__ v16h load_a16(const _Float16* A, int lda,
                                         int row0, int kk0, int lane) {
  const int hf = (lane >> 4) & 1, r = lane & 15;
  v16h a;
#pragma unroll
  for (int e = 0; e < 16; ++e) {
    int kk = kk0 + (hf << 4) + e;
    a[e] = A[(long long)(row0 + r) * lda + kk];
  }
  return a;
}

// B-matrix fragment: 32x16 f16 (K x N), row-major source, leading dim ldb.
// lane L holds col (L%16); element e -> K = kk0 + (L/16)*16 + e
__device__ __forceinline__ v16h load_b16(const _Float16* B, int ldb,
                                         int kk0, int col0, int lane) {
  const int hf = (lane >> 4) & 1, c = lane & 15;
  v16h b;
#pragma unroll
  for (int e = 0; e < 16; ++e)
    b[e] = B[(long long)(kk0 + (hf << 4) + e) * ldb + (col0 + c)];
  return b;
}

// B fragment where the logical B[kk][n] comes from a row-major matrix Bt[n][kk]
// (i.e. B = Bt^T).  Used for Q*K^T: contiguous in e.
__device__ __forceinline__ v16h load_bT16(const _Float16* Bt, int ldbt,
                                          int kk0, int col0, int lane) {
  const int hf = (lane >> 4) & 1, c = lane & 15;
  v16h b;
#pragma unroll
  for (int e = 0; e < 16; ++e)
    b[e] = Bt[(long long)(col0 + c) * ldbt + kk0 + (hf << 4) + e];
  return b;
}

// ---------------------------------------------------------------------------
// 1) per-proposal prep: size argmax + box, softmax prob/argmax, sort key
// ---------------------------------------------------------------------------
__global__ void prep_kernel(const float* __restrict__ size_scores,
                            const float* __restrict__ srn,
                            const float* __restrict__ sem,
                            const float* __restrict__ mean_size,
                            float* prob, int* label, float* box, float* skey) {
  int k = blockIdx.x * blockDim.x + threadIdx.x;
  if (k >= KP) return;
  int best = 0; float bv = size_scores[k * NS];
  for (int s = 1; s < NS; ++s) {
    float v = size_scores[k * NS + s];
    if (v > bv) { bv = v; best = s; }
  }
#pragma unroll
  for (int c = 0; c < 3; ++c)
    box[k * 3 + c] = mean_size[best * 3 + c] *
                     (1.0f + srn[(k * NS + best) * 3 + c]);
  float mx = sem[k * NC]; int am = 0;
  for (int j = 1; j < NC; ++j) {
    float v = sem[k * NC + j];
    if (v > mx) { mx = v; am = j; }
  }
  float sum = 0.0f;
  for (int j = 0; j < NC; ++j) sum += __expf(sem[k * NC + j] - mx);
  float p = 1.0f / sum;
  prob[k] = p;
  label[k] = am;
  skey[k] = (am > 0) ? p : -1.0f;
}

// ---------------------------------------------------------------------------
// 2) bitonic argsort (descending by key, tie -> smaller index), one block
// ---------------------------------------------------------------------------
__global__ void __launch_bounds__(1024)
sort_kernel(const float* __restrict__ skey, int* __restrict__ order) {
  __shared__ float sk[KP];
  __shared__ int   si[KP];
  for (int i = threadIdx.x; i < KP; i += blockDim.x) { sk[i] = skey[i]; si[i] = i; }
  __syncthreads();
  for (int sz = 2; sz <= KP; sz <<= 1) {
    for (int j = sz >> 1; j > 0; j >>= 1) {
      for (int i = threadIdx.x; i < KP; i += blockDim.x) {
        int ixj = i ^ j;
        if (ixj > i) {
          bool desc = ((i & sz) == 0);
          float a = sk[i], b = sk[ixj];
          int   ia = si[i], ib = si[ixj];
          bool a_first = (a > b) || (a == b && ia < ib);
          bool do_swap = desc ? !a_first : a_first;
          if (do_swap) { sk[i] = b; sk[ixj] = a; si[i] = ib; si[ixj] = ia; }
        }
      }
      __syncthreads();
    }
  }
  for (int i = threadIdx.x; i < KP; i += blockDim.x) order[i] = si[i];
}

// ---------------------------------------------------------------------------
// 3) gather sorted features (f32 + f16), write labels/bboxes/sorted_prob
// ---------------------------------------------------------------------------
__global__ void gather_kernel(const float* __restrict__ feats,
                              const float* __restrict__ prob,
                              const int* __restrict__ label,
                              const float* __restrict__ box,
                              const int* __restrict__ order,
                              float* sf32, _Float16* sf16, float* sprob,
                              float* out_labels, float* out_bbox) {
  int i = blockIdx.x;
  int src = order[i];
  for (int d = threadIdx.x; d < DD; d += blockDim.x) {
    float v = feats[(long long)src * DD + d];
    sf32[(long long)i * DD + d] = v;
    sf16[(long long)i * DD + d] = (_Float16)v;
  }
  if (threadIdx.x == 0) {
    int lb = label[src];
    sprob[i] = (lb > 0) ? prob[src] : 0.0f;
    out_labels[i] = (float)(lb - 1);
#pragma unroll
    for (int c = 0; c < 3; ++c) out_bbox[i * 3 + c] = box[src * 3 + c];
  }
}

// ---------------------------------------------------------------------------
// 4a) f32 -> f16 conversion (weights)
// ---------------------------------------------------------------------------
__global__ void cvt_kernel(const float* __restrict__ src, _Float16* dst, int n) {
  for (int i = blockIdx.x * blockDim.x + threadIdx.x; i < n;
       i += gridDim.x * blockDim.x)
    dst[i] = (_Float16)src[i];
}

// 4b) sinusoidal rank embedding, directly in f16
__global__ void emb_kernel(_Float16* emb16) {
  const int half = DD / 2;
  const float LN_WAVE = 6.907755278982137f;  // ln(1000)
  for (int idx = blockIdx.x * blockDim.x + threadIdx.x; idx < KP * DD;
       idx += gridDim.x * blockDim.x) {
    int k = idx / DD, j = idx % DD;
    int jj = (j < half) ? j : j - half;
    float inv = __expf(-((float)jj / (float)half) * LN_WAVE);
    float ang = (float)k * inv;
    emb16[idx] = (_Float16)((j < half) ? __sinf(ang) : __cosf(ang));
  }
}

// ---------------------------------------------------------------------------
// 5) e = emb @ w_rank + sf @ w_roi + b_rank + b_roi   -> f16 [KP, DF]
// ---------------------------------------------------------------------------
__global__ void e_kernel(const _Float16* __restrict__ emb16,
                         const _Float16* __restrict__ wr16,
                         const _Float16* __restrict__ sf16,
                         const _Float16* __restrict__ wo16,
                         const float* __restrict__ b_rank,
                         const float* __restrict__ b_roi,
                         _Float16* __restrict__ e16) {
  int wave = (blockIdx.x * blockDim.x + threadIdx.x) >> 5;
  int lane = threadIdx.x & 31;
  const int ntile = DF / 16;
  const int total = (KP / 16) * ntile;
  if (wave >= total) return;                 // wave-uniform guard
  int mt = wave / ntile, nt = wave % ntile;
  v8f acc = {};
  for (int kk = 0; kk < DD; kk += 32) {
    acc = wmma16(load_a16(emb16, DD, mt * 16, kk, lane),
                 load_b16(wr16, DF, kk, nt * 16, lane), acc);
    acc = wmma16(load_a16(sf16, DD, mt * 16, kk, lane),
                 load_b16(wo16, DF, kk, nt * 16, lane), acc);
  }
  int hf = lane >> 4, c = lane & 15;
  float bias = b_rank[nt * 16 + c] + b_roi[nt * 16 + c];
#pragma unroll
  for (int r = 0; r < 8; ++r)
    e16[(long long)(mt * 16 + hf * 8 + r) * DF + nt * 16 + c] =
        (_Float16)(acc[r] + bias);
}

// ---------------------------------------------------------------------------
// 6) batched f16 GEMM: C[b] = A[b] @ B[b], 16x16 tile per wave, f32 accum
// ---------------------------------------------------------------------------
__global__ void gemm16_kernel(const _Float16* __restrict__ A,
                              const _Float16* __restrict__ B,
                              _Float16* __restrict__ C,
                              int M, int N, int Kd,
                              long long sA, long long sB, long long sC,
                              int batch) {
  int lane = threadIdx.x & 31;
  int wave = (blockIdx.x * blockDim.x + threadIdx.x) >> 5;
  int nwaves = (gridDim.x * blockDim.x) >> 5;
  int tpb = (M / 16) * (N / 16);
  int total = batch * tpb;
  for (int t = wave; t < total; t += nwaves) {     // wave-uniform loop
    int b = t / tpb, tt = t % tpb;
    int mt = tt / (N / 16), nt = tt % (N / 16);
    const _Float16* Ab = A + (long long)b * sA;
    const _Float16* Bb = B + (long long)b * sB;
    v8f acc = {};
    for (int kk = 0; kk < Kd; kk += 32)
      acc = wmma16(load_a16(Ab, Kd, mt * 16, kk, lane),
                   load_b16(Bb, N, kk, nt * 16, lane), acc);
    int hf = lane >> 4, c = lane & 15;
    _Float16* Cb = C + (long long)b * sC;
#pragma unroll
    for (int r = 0; r < 8; ++r)
      Cb[(long long)(mt * 16 + hf * 8 + r) * N + nt * 16 + c] = (_Float16)acc[r];
  }
}

// ---------------------------------------------------------------------------
// 7) fused flash attention: one (head, 16-query-row tile) per wave.
//    K/V blocks (32 x 64 f16 = 4 KB each) are double-buffered in per-wave LDS
//    via GLOBAL_LOAD_ASYNC_TO_LDS (ASYNCcnt), overlapping the next block's
//    fetch with the current block's WMMAs + online softmax.
// ---------------------------------------------------------------------------
#define KVBLK (32 * DK)   // elements per staged K or V block

// stage one 32x64 f16 block (contiguous 4 KB): 8 async b128 ops per lane
__device__ __forceinline__ void stage_block(const _Float16* g, _Float16* l,
                                            int lane) {
#pragma unroll
  for (int i = 0; i < 8; ++i) {
    int off = (lane + 32 * i) * 8;   // in f16 elements (16 B chunks)
    cp16_g2l(g + off, l + off);
  }
}

__global__ void __launch_bounds__(128)
attn_kernel(const _Float16* __restrict__ q,
            const _Float16* __restrict__ kmat,
            const _Float16* __restrict__ v,
            float* __restrict__ o32) {
  const int lane = threadIdx.x & 31;
  const int wave = (blockIdx.x * blockDim.x + threadIdx.x) >> 5;
  const int wib  = threadIdx.x >> 5;       // wave-in-block (0..3)
  const int nqt = KP / 16;                 // 128 query tiles
  const int h = wave / nqt;
  const int qt = wave % nqt;
  if (h >= NH) return;                     // wave-uniform

  const _Float16* qh = q    + (long long)h * KP * DK;
  const _Float16* kh = kmat + (long long)h * KP * DK;
  const _Float16* vh = v    + (long long)h * KP * DK;

  // Q fragments for this 16-row tile (dk = 64 -> two 16x32 A frags)
  v16h qa0 = load_a16(qh, DK, qt * 16, 0, lane);
  v16h qa1 = load_a16(qh, DK, qt * 16, 32, lane);

  v8f acc[4] = {{}, {}, {}, {}};           // O tile: 16 x 64
  float m_i[8], l_i[8];
#pragma unroll
  for (int r = 0; r < 8; ++r) { m_i[r] = -1e30f; l_i[r] = 0.0f; }

  // per-wave LDS: double-buffered K/V blocks + P staging
  __shared__ _Float16 kstage[4][2][KVBLK];
  __shared__ _Float16 vstage[4][2][KVBLK];
  __shared__ _Float16 pstage[4][16 * 32];
  _Float16* ps = pstage[wib];

  const int hf = lane >> 4, c = lane & 15;
  const float scale = 0.125f;              // 1/sqrt(64)

  // preload first K/V block into buffer 0 (16 async ops)
  stage_block(kh, kstage[wib][0], lane);
  stage_block(vh, vstage[wib][0], lane);

  int cur = 0;
  for (int kb = 0; kb < KP; kb += 32) {
    if (kb + 32 < KP) {
      // issue next block into the other buffer, then wait until only those
      // 16 ops remain outstanding (in-order completion => current is ready)
      stage_block(kh + (long long)(kb + 32) * DK, kstage[wib][cur ^ 1], lane);
      stage_block(vh + (long long)(kb + 32) * DK, vstage[wib][cur ^ 1], lane);
      WAIT_ASYNC(16);
    } else {
      WAIT_ASYNC(0);
    }
    const _Float16* Kl = kstage[wib][cur];
    const _Float16* Vl = vstage[wib][cur];

    // S = Q K^T for key cols [kb, kb+32): two 16x16 tiles (K read from LDS)
    v8f s0 = {}, s1 = {};
    s0 = wmma16(qa0, load_bT16(Kl, DK, 0,  0,  lane), s0);
    s0 = wmma16(qa1, load_bT16(Kl, DK, 32, 0,  lane), s0);
    s1 = wmma16(qa0, load_bT16(Kl, DK, 0,  16, lane), s1);
    s1 = wmma16(qa1, load_bT16(Kl, DK, 32, 16, lane), s1);
#pragma unroll
    for (int r = 0; r < 8; ++r) { s0[r] *= scale; s1[r] *= scale; }

    // row max over the 32-col block (reduce across 16 lanes of this half)
    float mnew[8];
#pragma unroll
    for (int r = 0; r < 8; ++r) mnew[r] = fmaxf(s0[r], s1[r]);
    for (int off = 1; off < 16; off <<= 1)
#pragma unroll
      for (int r = 0; r < 8; ++r)
        mnew[r] = fmaxf(mnew[r], __shfl_xor(mnew[r], off, 32));

    float alpha[8];
#pragma unroll
    for (int r = 0; r < 8; ++r) {
      float m2 = fmaxf(m_i[r], mnew[r]);
      alpha[r] = __expf(m_i[r] - m2);
      m_i[r] = m2;
    }
    float rs[8];
#pragma unroll
    for (int r = 0; r < 8; ++r) {
      s0[r] = __expf(s0[r] - m_i[r]);
      s1[r] = __expf(s1[r] - m_i[r]);
      rs[r] = s0[r] + s1[r];
    }
    for (int off = 1; off < 16; off <<= 1)
#pragma unroll
      for (int r = 0; r < 8; ++r) rs[r] += __shfl_xor(rs[r], off, 32);
#pragma unroll
    for (int r = 0; r < 8; ++r) l_i[r] = l_i[r] * alpha[r] + rs[r];
#pragma unroll
    for (int t = 0; t < 4; ++t)
#pragma unroll
      for (int r = 0; r < 8; ++r) acc[t][r] *= alpha[r];

    // stage P (16x32 row-major f16) through LDS to re-fragment as A-matrix
#pragma unroll
    for (int r = 0; r < 8; ++r) {
      ps[(hf * 8 + r) * 32 + c]      = (_Float16)s0[r];
      ps[(hf * 8 + r) * 32 + 16 + c] = (_Float16)s1[r];
    }
    asm volatile("s_wait_dscnt 0x0" ::: "memory");   // wave-local LDS RAW
    v16h pa = load_a16(ps, 32, 0, 0, lane);

    // O += P @ V  (V block from LDS: 32 x 64 -> four 16-col tiles)
#pragma unroll
    for (int t = 0; t < 4; ++t)
      acc[t] = wmma16(pa, load_b16(Vl, DK, 0, t * 16, lane), acc[t]);

    cur ^= 1;
  }

  // normalize and scatter into o32[K, D] at head column offset
#pragma unroll
  for (int t = 0; t < 4; ++t)
#pragma unroll
    for (int r = 0; r < 8; ++r)
      o32[(long long)(qt * 16 + hf * 8 + r) * DD + h * DK + t * 16 + c] =
          acc[t][r] / l_i[r];
}

// ---------------------------------------------------------------------------
// 8) logit: sigmoid((sf + o) . w_logit + b) * sorted_prob
// ---------------------------------------------------------------------------
__global__ void __launch_bounds__(256)
logit_kernel(const float* __restrict__ sf32, const float* __restrict__ o32,
             const float* __restrict__ wl, const float* __restrict__ bl,
             const float* __restrict__ sprob, float* __restrict__ out_scores) {
  int i = blockIdx.x;
  __shared__ float red[256];
  float acc = 0.0f;
  for (int d = threadIdx.x; d < DD; d += blockDim.x)
    acc += (sf32[(long long)i * DD + d] + o32[(long long)i * DD + d]) * wl[d];
  red[threadIdx.x] = acc;
  __syncthreads();
  for (int s = 128; s > 0; s >>= 1) {
    if (threadIdx.x < s) red[threadIdx.x] += red[threadIdx.x + s];
    __syncthreads();
  }
  if (threadIdx.x == 0) {
    float lg = red[0] + bl[0];
    out_scores[i] = sprob[i] / (1.0f + __expf(-lg));
  }
}

// ---------------------------------------------------------------------------
extern "C" void kernel_launch(void* const* d_in, const int* in_sizes, int n_in,
                              void* d_out, int out_size, void* d_ws,
                              size_t ws_size, hipStream_t stream) {
  const float* size_scores = (const float*)d_in[0];
  const float* srn         = (const float*)d_in[1];
  const float* sem         = (const float*)d_in[2];
  const float* feats       = (const float*)d_in[3];
  const float* mean_size   = (const float*)d_in[4];
  const float* w_rank      = (const float*)d_in[5];
  const float* b_rank      = (const float*)d_in[6];
  const float* w_roi       = (const float*)d_in[7];
  const float* b_roi       = (const float*)d_in[8];
  const float* Wq          = (const float*)d_in[9];
  const float* Wk          = (const float*)d_in[10];
  const float* Wv          = (const float*)d_in[11];
  const float* w_logit     = (const float*)d_in[12];
  const float* b_logit     = (const float*)d_in[13];

  float* out_scores = (float*)d_out;          // [K]
  float* out_labels = out_scores + KP;        // [K] (label-1, as float)
  float* out_bbox   = out_scores + 2 * KP;    // [K,3]

  char* base = (char*)d_ws;
  size_t off = 0;
  auto alloc = [&](size_t bytes) -> void* {
    void* p = base + off;
    off += (bytes + 255) & ~(size_t)255;
    return p;
  };

  float*     prob   = (float*)alloc(KP * 4);
  int*       label  = (int*)alloc(KP * 4);
  float*     box    = (float*)alloc(KP * 3 * 4);
  float*     skey   = (float*)alloc(KP * 4);
  int*       order  = (int*)alloc(KP * 4);
  float*     sprob  = (float*)alloc(KP * 4);
  float*     sf32   = (float*)alloc((size_t)KP * DD * 4);
  _Float16*  sf16   = (_Float16*)alloc((size_t)KP * DD * 2);
  _Float16*  emb16  = (_Float16*)alloc((size_t)KP * DD * 2);
  _Float16*  wr16   = (_Float16*)alloc((size_t)DD * DF * 2);
  _Float16*  wo16   = (_Float16*)alloc((size_t)DD * DF * 2);
  _Float16*  wq16   = (_Float16*)alloc((size_t)NH * DF * DK * 2);
  _Float16*  wk16   = (_Float16*)alloc((size_t)NH * DF * DK * 2);
  _Float16*  wv16   = (_Float16*)alloc((size_t)NH * DD * DK * 2);
  _Float16*  e16    = (_Float16*)alloc((size_t)KP * DF * 2);
  _Float16*  q16    = (_Float16*)alloc((size_t)NH * KP * DK * 2);
  _Float16*  k16    = (_Float16*)alloc((size_t)NH * KP * DK * 2);
  _Float16*  v16    = (_Float16*)alloc((size_t)NH * KP * DK * 2);
  float*     o32    = (float*)alloc((size_t)KP * DD * 4);
  (void)ws_size; (void)in_sizes; (void)n_in; (void)out_size;

  prep_kernel<<<(KP + 255) / 256, 256, 0, stream>>>(
      size_scores, srn, sem, mean_size, prob, label, box, skey);
  sort_kernel<<<1, 1024, 0, stream>>>(skey, order);
  gather_kernel<<<KP, 256, 0, stream>>>(feats, prob, label, box, order,
                                        sf32, sf16, sprob, out_labels, out_bbox);
  cvt_kernel<<<256, 256, 0, stream>>>(w_rank, wr16, DD * DF);
  cvt_kernel<<<256, 256, 0, stream>>>(w_roi,  wo16, DD * DF);
  cvt_kernel<<<256, 256, 0, stream>>>(Wq, wq16, NH * DF * DK);
  cvt_kernel<<<256, 256, 0, stream>>>(Wk, wk16, NH * DF * DK);
  cvt_kernel<<<1024, 256, 0, stream>>>(Wv, wv16, NH * DD * DK);
  emb_kernel<<<2048, 256, 0, stream>>>(emb16);
  e_kernel<<<256, 128, 0, stream>>>(emb16, wr16, sf16, wo16, b_rank, b_roi, e16);
  gemm16_kernel<<<2048, 128, 0, stream>>>(e16, wq16, q16, KP, DK, DF,
                                          0LL, (long long)DF * DK,
                                          (long long)KP * DK, NH);
  gemm16_kernel<<<2048, 128, 0, stream>>>(e16, wk16, k16, KP, DK, DF,
                                          0LL, (long long)DF * DK,
                                          (long long)KP * DK, NH);
  gemm16_kernel<<<2048, 128, 0, stream>>>(sf16, wv16, v16, KP, DK, DD,
                                          0LL, (long long)DD * DK,
                                          (long long)KP * DK, NH);
  attn_kernel<<<512, 128, 0, stream>>>(q16, k16, v16, o32);
  logit_kernel<<<KP, 256, 0, stream>>>(sf32, o32, w_logit, b_logit, sprob,
                                       out_scores);
}
